// PhysNetmodel_77017353552179
// MI455X (gfx1250) — matrix operand order repaired and checked by reading, not compiled, not run on hardware
//
#include <hip/hip_runtime.h>

// ---------------------------------------------------------------------------
// PhysNet forward for MI455X (gfx1250, wave32, WMMA).
// - All 128x128 dense layers + the [E,64]@[64,128] edge gate GEMM run on
//   v_wmma_f32_16x16x32_bf16.
// - Weights converted to bf16 *pre-swizzled into fragment order*: each lane's
//   B fragment is one contiguous 32-byte LDS read (2x ds_load_b128).
// - A fragments are two contiguous 16-byte global loads per lane.
// - The [E,128] gate tensor g is never materialized (fused GEMM+gather+scatter,
//   f32 atomics over sorted idx_i). All gather/scatter addressing uses
//   uniform SGPR bases + 32-bit unsigned VGPR offsets (SADDR/GVS form) with
//   compile-time immediate offsets -> no 64-bit address VALU in the hot loop.
// - ssp(.)->bf16 for the next layer is fused into the GEMM epilogue.
// ---------------------------------------------------------------------------

#define F 128
#define KD 64
#define NA 50000
#define NE 800000
#define NM 64
#define SRCUT 10.0f
#define KEHALF 7.199822675975274f
#define LN2F 0.6931471824645996f

typedef __bf16 bf16;
typedef __attribute__((ext_vector_type(16))) __bf16 v16bf;
typedef __attribute__((ext_vector_type(8)))  float  v8f;

__device__ __forceinline__ bf16 f2bf(float f) { return (bf16)f; }

__device__ __forceinline__ float sspf(float x) {
  // shifted softplus, numerically stable: max(x,0)+log1p(exp(-|x|)) - ln2
  float ax = fabsf(x);
  return fmaxf(x, 0.0f) + __logf(1.0f + __expf(-ax)) - LN2F;
}

// --- WMMA fragment loaders (CDNA5 16-bit layouts, wave32) -------------------
// A 16x32 fragment of a row-major [rows x ld] bf16 matrix. Per-lane data is
// two contiguous 16B runs: k = kb+[0..7] and kb+[16..23], kb = koff + 8*(lane>=16).
__device__ __forceinline__ v16bf frag_a(const bf16* __restrict__ base, int ld,
                                        int lane, int koff) {
  const bf16* rp = base + (size_t)(lane & 15) * ld + koff + ((lane < 16) ? 0 : 8);
  union { v16bf v; uint4 q[2]; } u;
  u.q[0] = *(const uint4*)(rp);
  u.q[1] = *(const uint4*)(rp + 16);
  return u.v;
}

// B 32x16 fragment from a pre-swizzled weight buffer: fragment `frag`
// (= kc*8 + nt), lane data contiguous 32 bytes.
__device__ __forceinline__ v16bf frag_b_swz(const bf16* __restrict__ sW,
                                            int frag, int lane) {
  const uint4* p = (const uint4*)(sW + ((size_t)frag * 32 + lane) * 16);
  union { v16bf v; uint4 q[2]; } u;
  u.q[0] = p[0];
  u.q[1] = p[1];
  return u.v;
}

// --- Dense layer: out[N,128] = act(A_bf16[N,128] @ W[128,128] + b (+addend))
// Optionally also writes bf16(out) and/or bf16(ssp(out)) for the next layer.
// One wave per 16-row tile; 8 waves/workgroup; swizzled W staged in LDS (32KB).
template <bool ADD, bool ACT, bool WBF, bool WSSP>
__global__ void __launch_bounds__(256) pn_gemm(
    const bf16* __restrict__ A, const bf16* __restrict__ Wswz,
    const float* __restrict__ bias, const float* __restrict__ addend,
    float* __restrict__ out, bf16* __restrict__ out_bf,
    bf16* __restrict__ out_ssp, int nrows) {
  __shared__ __align__(16) bf16 sW[F * F];
  int tid = threadIdx.x;
  {
    const uint4* src = (const uint4*)Wswz;
    uint4* dst = (uint4*)sW;
    for (int i = tid; i < F * F / 8; i += 256) dst[i] = src[i];
  }
  __syncthreads();

  int lane = tid & 31;
  int row0 = (blockIdx.x * 8 + (tid >> 5)) * 16;
  if (row0 >= nrows) return;

  const bf16* Ab = A + (size_t)row0 * F;
  v16bf a0 = frag_a(Ab, F, lane, 0);
  v16bf a1 = frag_a(Ab, F, lane, 32);
  v16bf a2 = frag_a(Ab, F, lane, 64);
  v16bf a3 = frag_a(Ab, F, lane, 96);

  unsigned lane15 = lane & 15;
  unsigned mb = (lane < 16) ? 0u : 8u;
  // single 32-bit base offset; all element accesses below are base + constant
  unsigned boff = (unsigned)(row0 + mb) * F + lane15;

#pragma unroll
  for (int nt = 0; nt < 8; ++nt) {
    v8f c = {};
    c = __builtin_amdgcn_wmma_f32_16x16x32_bf16(false, a0, false,
          frag_b_swz(sW, 0 * 8 + nt, lane), (short)0, c, false, false);
    c = __builtin_amdgcn_wmma_f32_16x16x32_bf16(false, a1, false,
          frag_b_swz(sW, 1 * 8 + nt, lane), (short)0, c, false, false);
    c = __builtin_amdgcn_wmma_f32_16x16x32_bf16(false, a2, false,
          frag_b_swz(sW, 2 * 8 + nt, lane), (short)0, c, false, false);
    c = __builtin_amdgcn_wmma_f32_16x16x32_bf16(false, a3, false,
          frag_b_swz(sW, 3 * 8 + nt, lane), (short)0, c, false, false);
    float bn = bias[lane15 + nt * 16];
#pragma unroll
    for (int v = 0; v < 8; ++v) {
      unsigned o = boff + (unsigned)(v * F + nt * 16);  // imm-foldable
      float val = c[v] + bn;
      if (ADD) val += addend[o];
      if (ACT) val = sspf(val);
      out[o] = val;
      if (WBF)  out_bf[o] = f2bf(val);
      if (WSSP) out_ssp[o] = f2bf(sspf(val));
    }
  }
}

// --- Fused edge messages: g = rbf@Wg (WMMA), vj = g * vjb[idx_j],
//     mbuf[idx_i] += vj (f32 atomics; idx_i sorted => L2-friendly).
__global__ void __launch_bounds__(256) pn_edge_msg(
    const bf16* __restrict__ rbf, const bf16* __restrict__ Wgswz,
    const float* __restrict__ vjb, const int* __restrict__ ii,
    const int* __restrict__ jj, float* __restrict__ mbuf, int nedges) {
  __shared__ __align__(16) bf16 sW[KD * F];   // 16 KB
  int tid = threadIdx.x;
  {
    const uint4* src = (const uint4*)Wgswz;
    uint4* dst = (uint4*)sW;
    for (int i = tid; i < KD * F / 8; i += 256) dst[i] = src[i];
  }
  __syncthreads();

  int lane = tid & 31;
  int e0 = (blockIdx.x * 8 + (tid >> 5)) * 16;
  if (e0 >= nedges) return;

  const bf16* Ab = rbf + (size_t)e0 * KD;
  v16bf a0 = frag_a(Ab, KD, lane, 0);
  v16bf a1 = frag_a(Ab, KD, lane, 32);

  unsigned lane15 = lane & 15;
  int mb = (lane < 16) ? 0 : 8;
  // 32-bit unsigned element offsets per row (SADDR + voffset + imm form).
  unsigned oj[8], oi[8];
#pragma unroll
  for (int v = 0; v < 8; ++v) {
    oj[v] = (unsigned)jj[e0 + mb + v] * F + lane15;
    oi[v] = (unsigned)ii[e0 + mb + v] * F + lane15;
  }

#pragma unroll
  for (int nt = 0; nt < 8; ++nt) {
    v8f c = {};
    c = __builtin_amdgcn_wmma_f32_16x16x32_bf16(false, a0, false,
          frag_b_swz(sW, 0 * 8 + nt, lane), (short)0, c, false, false);
    c = __builtin_amdgcn_wmma_f32_16x16x32_bf16(false, a1, false,
          frag_b_swz(sW, 1 * 8 + nt, lane), (short)0, c, false, false);
    float vj[8];
#pragma unroll
    for (int v = 0; v < 8; ++v) vj[v] = vjb[oj[v] + (unsigned)(nt * 16)];
#pragma unroll
    for (int v = 0; v < 8; ++v)
      atomicAdd(&mbuf[oi[v] + (unsigned)(nt * 16)], c[v] * vj[v]);
  }
}

// --- Weight conversion f32 -> bf16, swizzled to WMMA fragment order ---------
// dst[((kc*8+nt)*32 + lane)*16 + e] = W[(kc*32 + 8*(lane>=16) + (e&7) + 16*(e>=8))*128 + nt*16 + lane%16]
__global__ void pn_f2bf_swz(const float* __restrict__ W, bf16* __restrict__ dst,
                            int n) {  // n = K*128
  int i = blockIdx.x * blockDim.x + threadIdx.x;
  if (i >= n) return;
  int e = i & 15;
  int lane = (i >> 4) & 31;
  int frag = i >> 9;
  int k = (frag >> 3) * 32 + ((lane < 16) ? 0 : 8) + (e & 7) + ((e >> 3) << 4);
  int col = (frag & 7) * 16 + (lane & 15);
  dst[i] = f2bf(W[(size_t)k * F + col]);
}

// --- Small elementwise / edge / reduction kernels ---------------------------
__global__ void pn_ssp_bf(const float* __restrict__ s, bf16* __restrict__ d, int n) {
  int i = blockIdx.x * blockDim.x + threadIdx.x;
  if (i < n) d[i] = f2bf(sspf(s[i]));
}

__global__ void pn_zero(float* __restrict__ p, int n) {
  int i = blockIdx.x * blockDim.x + threadIdx.x;
  if (i < n) p[i] = 0.0f;
}

__global__ void pn_mulcol(const float* __restrict__ x, const float* __restrict__ u,
                          float* __restrict__ out, int n) {
  int i = blockIdx.x * blockDim.x + threadIdx.x;
  if (i < n) out[i] = x[i] * u[i & (F - 1)];
}

__global__ void pn_init_x(const int* __restrict__ Z, const float* __restrict__ emb,
                          float* __restrict__ x) {
  int i = blockIdx.x * blockDim.x + threadIdx.x;
  if (i >= NA * F) return;
  int a = i >> 7, n = i & (F - 1);
  x[i] = emb[(size_t)Z[a] * F + n];
}

__global__ void pn_edge_geom(const float* __restrict__ R, const int* __restrict__ ii,
                             const int* __restrict__ jj,
                             const float* __restrict__ cent,
                             const float* __restrict__ wid,
                             float* __restrict__ Dij, bf16* __restrict__ rbf) {
  int e = blockIdx.x * blockDim.x + threadIdx.x;
  if (e >= NE) return;
  int i = ii[e], j = jj[e];
  float dx = R[i * 3 + 0] - R[j * 3 + 0];
  float dy = R[i * 3 + 1] - R[j * 3 + 1];
  float dz = R[i * 3 + 2] - R[j * 3 + 2];
  float D = sqrtf(fmaxf(dx * dx + dy * dy + dz * dz, 0.0f));
  Dij[e] = D;
  float xc = D / SRCUT;
  float cutoff = (D < SRCUT) ? 1.0f - ((6.0f * xc - 15.0f) * xc + 10.0f) * xc * xc * xc
                             : 0.0f;
  float ed = __expf(-D);
  bf16* out = rbf + (size_t)e * KD;
#pragma unroll 4
  for (int k = 0; k < KD; ++k) {
    float t = ed - cent[k];
    out[k] = f2bf(cutoff * __expf(-wid[k] * t * t));
  }
}

__global__ void pn_out_proj(const float* __restrict__ o, const float* __restrict__ Wod,
                            const float* __restrict__ bod,
                            float* __restrict__ Ea, float* __restrict__ Qa) {
  int a = blockIdx.x * blockDim.x + threadIdx.x;
  if (a >= NA) return;
  const float* row = o + (size_t)a * F;
  float e = bod[0], q = bod[1];
#pragma unroll 4
  for (int k = 0; k < F; ++k) {
    float s = sspf(row[k]);
    e += s * Wod[k * 2 + 0];
    q += s * Wod[k * 2 + 1];
  }
  Ea[a] += e;
  Qa[a] += q;
}

__global__ void pn_escale(const int* __restrict__ Z,
                          const float* __restrict__ Es, const float* __restrict__ Ef,
                          const float* __restrict__ Qs, const float* __restrict__ Qf,
                          float* __restrict__ Ea, float* __restrict__ Qa) {
  int a = blockIdx.x * blockDim.x + threadIdx.x;
  if (a >= NA) return;
  int z = Z[a];
  Ea[a] = Es[z] * Ea[a] + Ef[z];
  Qa[a] = Qs[z] * Qa[a] + Qf[z];
}

__global__ void pn_mol_acc(const float* __restrict__ Qa, const int* __restrict__ seg,
                           float* __restrict__ Na, float* __restrict__ Qsum) {
  int a = blockIdx.x * blockDim.x + threadIdx.x;
  if (a >= NA) return;
  int s = seg[a];
  atomicAdd(&Na[s], 1.0f);
  atomicAdd(&Qsum[s], Qa[a]);
}

__global__ void pn_charge_fix(float* __restrict__ Qa, const int* __restrict__ seg,
                              const float* __restrict__ Qtot,
                              const float* __restrict__ Na,
                              const float* __restrict__ Qsum) {
  int a = blockIdx.x * blockDim.x + threadIdx.x;
  if (a >= NA) return;
  int s = seg[a];
  Qa[a] += (Qtot[s] - Qsum[s]) / Na[s];
}

__global__ void pn_electro(const float* __restrict__ Qa, const float* __restrict__ Dij,
                           const int* __restrict__ ii, const int* __restrict__ jj,
                           float* __restrict__ Ea) {
  int e = blockIdx.x * blockDim.x + threadIdx.x;
  if (e >= NE) return;
  float D = Dij[e];
  float DS = sqrtf(D * D + 1.0f);
  float cut = SRCUT * 0.5f;
  float xs = D / cut;
  float sw = (D < cut) ? ((6.0f * xs - 15.0f) * xs + 10.0f) * xs * xs * xs : 1.0f;
  float val = KEHALF * Qa[ii[e]] * Qa[jj[e]] * ((1.0f - sw) / DS + sw / D);
  atomicAdd(&Ea[ii[e]], val);
}

__global__ void pn_seg_sum(const float* __restrict__ Ea, const int* __restrict__ seg,
                           float* __restrict__ out) {
  int a = blockIdx.x * blockDim.x + threadIdx.x;
  if (a >= NA) return;
  atomicAdd(&out[seg[a]], Ea[a]);
}

// ---------------------------------------------------------------------------
extern "C" void kernel_launch(void* const* d_in, const int* in_sizes, int n_in,
                              void* d_out, int out_size, void* d_ws, size_t ws_size,
                              hipStream_t stream) {
  (void)in_sizes; (void)n_in; (void)out_size; (void)ws_size;

  const int*   Z      = (const int*)d_in[0];
  const float* R      = (const float*)d_in[1];
  const int*   ii     = (const int*)d_in[2];
  const int*   jj     = (const int*)d_in[3];
  const int*   seg    = (const int*)d_in[4];
  const float* Qtot   = (const float*)d_in[5];
  const float* emb    = (const float*)d_in[6];
  const float* cent   = (const float*)d_in[7];
  const float* wid    = (const float*)d_in[8];
  const float* Eshift = (const float*)d_in[9];
  const float* Escale = (const float*)d_in[10];
  const float* Qshift = (const float*)d_in[11];
  const float* Qscale = (const float*)d_in[12];

  char* ws = (char*)d_ws;
  size_t off = 0;
  auto alloc = [&](size_t bytes) {
    void* p = ws + off;
    off += (bytes + 255) & ~(size_t)255;
    return p;
  };
  float* Dij   = (float*)alloc((size_t)NE * 4);
  bf16*  rbf   = (bf16*) alloc((size_t)NE * KD * 2);
  float* x     = (float*)alloc((size_t)NA * F * 4);
  float* m     = (float*)alloc((size_t)NA * F * 4);
  float* t0    = (float*)alloc((size_t)NA * F * 4);
  float* vjb   = (float*)alloc((size_t)NA * F * 4);
  float* o     = (float*)alloc((size_t)NA * F * 4);
  bf16*  xa_bf = (bf16*) alloc((size_t)NA * F * 2);
  bf16*  h_bf  = (bf16*) alloc((size_t)NA * F * 2);
  float* Ea    = (float*)alloc((size_t)NA * 4);
  float* Qa    = (float*)alloc((size_t)NA * 4);
  float* Na    = (float*)alloc(NM * 4);
  float* Qsum  = (float*)alloc(NM * 4);

  // Per-block parameter leaf offsets (insertion-order flattening of params).
  // role: 0 dense_i, 1 dense_j, 2..5 res_inter, 6 dense_out, 7..10 res_atomic,
  //       11..12 out_res
  const int dW[13] = {1, 3, 5, 7, 9, 11, 13, 16, 18, 20, 22, 24, 26};
  const int dB[13] = {2, 4, 6, 8, 10, 12, 14, 17, 19, 21, 23, 25, 27};

  bf16* Wg_bf[3];
  bf16* Wd_bf[3][13];
  for (int b = 0; b < 3; ++b) {
    Wg_bf[b] = (bf16*)alloc((size_t)KD * F * 2);
    for (int l = 0; l < 13; ++l) Wd_bf[b][l] = (bf16*)alloc((size_t)F * F * 2);
  }

  auto grid = [](long n, int blk) { return (unsigned)((n + blk - 1) / blk); };
  const long NF = (long)NA * F;
  const unsigned GG = grid(NA / 16, 8);   // gemm workgroups (8 waves x 16 rows)
  const float* nf = nullptr;

  // -- weight conversion to swizzled bf16 (deterministic, every call) --
  for (int b = 0; b < 3; ++b) {
    int base = 13 + b * 30;
    pn_f2bf_swz<<<grid(KD * F, 256), 256, 0, stream>>>(
        (const float*)d_in[base + 0], Wg_bf[b], KD * F);
    for (int l = 0; l < 13; ++l)
      pn_f2bf_swz<<<grid(F * F, 256), 256, 0, stream>>>(
          (const float*)d_in[base + dW[l]], Wd_bf[b][l], F * F);
  }

  // -- geometry / RBF / embeddings / accumulators --
  pn_edge_geom<<<grid(NE, 256), 256, 0, stream>>>(R, ii, jj, cent, wid, Dij, rbf);
  pn_init_x<<<grid(NF, 256), 256, 0, stream>>>(Z, emb, x);
  pn_zero<<<grid(NA, 256), 256, 0, stream>>>(Ea, NA);
  pn_zero<<<grid(NA, 256), 256, 0, stream>>>(Qa, NA);

  // xa_bf = bf16(ssp(x)) for the first block; later blocks get it fused from
  // the res_atomic[1] GEMM epilogue.
  pn_ssp_bf<<<grid(NF, 256), 256, 0, stream>>>(x, xa_bf, (int)NF);

  for (int b = 0; b < 3; ++b) {
    int base = 13 + b * 30;
    auto P = [&](int k) { return (const float*)d_in[base + k]; };

    // m = vi = ssp(dense_i(xa)) ; vjb = ssp(dense_j(xa))
    pn_gemm<false, true, false, false><<<GG, 256, 0, stream>>>(
        xa_bf, Wd_bf[b][0], P(dB[0]), nf, m, nullptr, nullptr, NA);
    pn_gemm<false, true, false, false><<<GG, 256, 0, stream>>>(
        xa_bf, Wd_bf[b][1], P(dB[1]), nf, vjb, nullptr, nullptr, NA);
    // m += scatter_i( (rbf@Wg) * vjb[idx_j] )  (fused WMMA + gather + atomics)
    pn_edge_msg<<<NE / 128, 256, 0, stream>>>(rbf, Wg_bf[b], vjb, ii, jj, m, NE);
    // xa_bf = bf16(ssp(m))
    pn_ssp_bf<<<grid(NF, 256), 256, 0, stream>>>(m, xa_bf, (int)NF);

    // res_inter x2 on m (gemm2 re-emits ssp(m) in bf16 for the next layer)
    for (int r = 0; r < 2; ++r) {
      int l1 = 2 + 2 * r, l2 = 3 + 2 * r;
      pn_gemm<false, true, true, false><<<GG, 256, 0, stream>>>(
          xa_bf, Wd_bf[b][l1], P(dB[l1]), nf, t0, h_bf, nullptr, NA);
      pn_gemm<true, false, false, true><<<GG, 256, 0, stream>>>(
          h_bf, Wd_bf[b][l2], P(dB[l2]), m, m, nullptr, xa_bf, NA);
    }

    // x = u*x + dense_out(ssp(m))   (xa_bf currently = ssp(m))
    pn_mulcol<<<grid(NF, 256), 256, 0, stream>>>(x, P(15), t0, (int)NF);
    pn_gemm<true, false, false, true><<<GG, 256, 0, stream>>>(
        xa_bf, Wd_bf[b][6], P(dB[6]), t0, x, nullptr, xa_bf, NA);

    // res_atomic x2 on x (each gemm2 re-emits ssp(x) in bf16)
    for (int r = 0; r < 2; ++r) {
      int l1 = 7 + 2 * r, l2 = 8 + 2 * r;
      pn_gemm<false, true, true, false><<<GG, 256, 0, stream>>>(
          xa_bf, Wd_bf[b][l1], P(dB[l1]), nf, t0, h_bf, nullptr, NA);
      pn_gemm<true, false, false, true><<<GG, 256, 0, stream>>>(
          h_bf, Wd_bf[b][l2], P(dB[l2]), x, x, nullptr, xa_bf, NA);
    }

    // out_res: o = residual(x); x and xa_bf(=ssp(x)) preserved for next block
    pn_gemm<false, true, true, false><<<GG, 256, 0, stream>>>(
        xa_bf, Wd_bf[b][11], P(dB[11]), nf, t0, h_bf, nullptr, NA);
    pn_gemm<true, false, false, false><<<GG, 256, 0, stream>>>(
        h_bf, Wd_bf[b][12], P(dB[12]), x, o, nullptr, nullptr, NA);

    // Ea/Qa += out_dense(ssp(o))
    pn_out_proj<<<grid(NA, 256), 256, 0, stream>>>(
        o, (const float*)d_in[base + 28], (const float*)d_in[base + 29], Ea, Qa);
  }

  // per-element scale/shift
  pn_escale<<<grid(NA, 256), 256, 0, stream>>>(Z, Escale, Eshift, Qscale, Qshift,
                                               Ea, Qa);
  // charge conservation
  pn_zero<<<1, 64, 0, stream>>>(Na, NM);
  pn_zero<<<1, 64, 0, stream>>>(Qsum, NM);
  pn_mol_acc<<<grid(NA, 256), 256, 0, stream>>>(Qa, seg, Na, Qsum);
  pn_charge_fix<<<grid(NA, 256), 256, 0, stream>>>(Qa, seg, Qtot, Na, Qsum);
  // shielded/switched electrostatics
  pn_electro<<<grid(NE, 256), 256, 0, stream>>>(Qa, Dij, ii, jj, Ea);
  // per-molecule energies
  pn_zero<<<1, 64, 0, stream>>>((float*)d_out, NM);
  pn_seg_sum<<<grid(NA, 256), 256, 0, stream>>>(Ea, seg, (float*)d_out);
}
